// BitLinear_8632884265300
// MI455X (gfx1250) — matrix operand order repaired and checked by reading, not compiled, not run on hardware
//
#include <hip/hip_runtime.h>
#include <hip/hip_bf16.h>
#include <math.h>

// ---------------------------------------------------------------------------
// BitLinear (1.58-bit weights): LayerNorm -> absmax act-quant -> ternary W
//   -> GEMM (32768x1024 @ 1024x1024) -> rescale.
// CDNA5 / gfx1250: wave32, v_wmma_f32_16x16x32_f16 (f32 acc),
// B tiles moved with global_load_async_to_lds_b128 (ASYNCcnt path).
// ---------------------------------------------------------------------------

typedef __attribute__((ext_vector_type(16))) _Float16 v16h;
typedef __attribute__((ext_vector_type(8)))  _Float16 v8h;
typedef __attribute__((ext_vector_type(8)))  float    v8f;
typedef __attribute__((ext_vector_type(4)))  float    v4f;

#define MROWS   32768            // 4 * 8192
#define NCOLS   1024
#define KDIM    1024

#define BM      128              // M tile per workgroup
#define BN      256              // N tile per workgroup
#define BK      32               // K step per stage (one wmma K)
#define ASTRIDE 40               // halfs per A row in LDS (pad 32->40, 80B rows)
#define BSTRIDE 40               // halfs per B row in LDS (16B-aligned rows)
#define NSTAGE  (KDIM / BK)      // 32

// workspace layout (float indices)
#define WS_BETA      0
#define WS_GAMMABITS 1           // uint bits of running absmax (values >= 0, monotonic)
#define WS_WSCALE    2
#define WS_QSCALE    3
#define WS_OUTSCALE  4
#define WS_WPART     8           // 256 partial |W| sums
#define WS_ROWSTATS  512         // 2*MROWS floats: (mu, rsigma) per row
#define WS_BTQ       66048       // pre-quantized transposed weights: f16 Bt[1024][1024]

union HPack { _Float16 h[2]; unsigned int u; };

// --------------------------- small kernels ---------------------------------

__global__ void init_kernel(float* ws) {
    if (threadIdx.x == 0) {
        ((unsigned int*)ws)[WS_GAMMABITS] = 0u;   // absmax accumulator
    }
}

// partial sums of |W| : deterministic (fixed tree, fixed partial order)
__global__ __launch_bounds__(256) void wpartial_kernel(const float* __restrict__ w,
                                                       float* __restrict__ ws) {
    __shared__ float red[256];
    const int t = threadIdx.x;
    const int base = blockIdx.x * 256 + t;
    float s = 0.0f;
    #pragma unroll
    for (int i = 0; i < 16; ++i)                   // 256 blocks * 256 thr * 16 = 1M
        s += fabsf(w[base + i * 65536]);
    red[t] = s;
    __syncthreads();
    for (int off = 128; off > 0; off >>= 1) {
        if (t < off) red[t] += red[t + off];
        __syncthreads();
    }
    if (t == 0) ws[WS_WPART + blockIdx.x] = red[0];
}

// per-row LayerNorm stats + global absmax of normalized activations
__global__ __launch_bounds__(256) void rowstats_kernel(const float* __restrict__ x,
                                                       const float* __restrict__ lng,
                                                       const float* __restrict__ lnb,
                                                       float* __restrict__ ws) {
    __shared__ float s1[256];
    __shared__ float s2[256];
    __shared__ float mu_s, rs_s;
    const int row = blockIdx.x;
    const int t   = threadIdx.x;

    v4f xv = *(const v4f*)(x + (size_t)row * KDIM + t * 4);
    float sum = xv[0] + xv[1] + xv[2] + xv[3];
    float sq  = xv[0]*xv[0] + xv[1]*xv[1] + xv[2]*xv[2] + xv[3]*xv[3];
    s1[t] = sum; s2[t] = sq;
    __syncthreads();
    for (int off = 128; off > 0; off >>= 1) {
        if (t < off) { s1[t] += s1[t + off]; s2[t] += s2[t + off]; }
        __syncthreads();
    }
    if (t == 0) {
        float mu  = s1[0] * (1.0f / KDIM);
        float var = s2[0] * (1.0f / KDIM) - mu * mu;
        float rs  = rsqrtf(var + 1e-3f);           // keras LN eps
        ws[WS_ROWSTATS + 2 * row]     = mu;
        ws[WS_ROWSTATS + 2 * row + 1] = rs;
        mu_s = mu; rs_s = rs;
    }
    __syncthreads();
    const float mu = mu_s, rs = rs_s;
    v4f gv = *(const v4f*)(lng + t * 4);
    v4f bv = *(const v4f*)(lnb + t * 4);
    float amax = 0.0f;
    #pragma unroll
    for (int e = 0; e < 4; ++e) {
        float v = (xv[e] - mu) * rs * gv[e] + bv[e];
        amax = fmaxf(amax, fabsf(v));
    }
    s1[t] = amax;
    __syncthreads();
    for (int off = 128; off > 0; off >>= 1) {
        if (t < off) s1[t] = fmaxf(s1[t], s1[t + off]);
        __syncthreads();
    }
    if (t == 0)  // max is order-independent -> deterministic
        atomicMax((unsigned int*)ws + WS_GAMMABITS, __float_as_uint(s1[0]));
}

__global__ __launch_bounds__(256) void finalize_kernel(float* __restrict__ ws) {
    __shared__ float red[256];
    const int t = threadIdx.x;
    red[t] = ws[WS_WPART + t];
    __syncthreads();
    for (int off = 128; off > 0; off >>= 1) {
        if (t < off) red[t] += red[t + off];
        __syncthreads();
    }
    if (t == 0) {
        float beta  = red[0] * (1.0f / (1024.0f * 1024.0f));   // mean |W|
        float gamma = __uint_as_float(((unsigned int*)ws)[WS_GAMMABITS]);
        ws[WS_BETA]     = beta;
        ws[WS_WSCALE]   = 1.0f / (beta + 1e-5f);
        ws[WS_QSCALE]   = 128.0f / gamma;
        ws[WS_OUTSCALE] = gamma * beta * (1.0f / 128.0f);
    }
}

// Quantize weights once: W(d,u) f32 -> ternary f16, stored TRANSPOSED as
// Bt[u][d] so the GEMM B-side becomes a pure async tile copy. LDS-tiled so
// both the global read and the global write are coalesced.
__global__ __launch_bounds__(256) void wquant_kernel(const float* __restrict__ w,
                                                     const float* __restrict__ ws,
                                                     _Float16* __restrict__ btq) {
    __shared__ _Float16 sh[64][72];               // 72-half stride: 16B-aligned rows
    const int t  = threadIdx.x;
    const int u0 = blockIdx.x * 64;
    const int d0 = blockIdx.y * 64;
    const float wscale = ws[WS_WSCALE];

    const int lu = t & 63;                        // column within tile
    const int g4 = t >> 6;                        // 0..3 -> 16-row slab
    #pragma unroll
    for (int r = 0; r < 16; ++r) {
        const int d = d0 + g4 * 16 + r;
        float wv = w[(size_t)d * NCOLS + u0 + lu];   // coalesced read
        float q  = rintf(wv * wscale);               // round-to-even (STE fwd)
        q = fminf(fmaxf(q, -1.0f), 1.0f);            // ternary clip
        sh[lu][g4 * 16 + r] = (_Float16)q;           // transpose in LDS
    }
    __syncthreads();
    const int chunk = t >> 6;                     // 0..3 -> 16 halfs along d
    v8h a = *(const v8h*)&sh[lu][chunk * 16];
    v8h b = *(const v8h*)&sh[lu][chunk * 16 + 8];
    _Float16* dst = btq + (size_t)(u0 + lu) * KDIM + d0 + chunk * 16;
    *(v8h*)dst       = a;                         // coalesced 16B writes
    *(v8h*)(dst + 8) = b;
}

// ------------------------------ GEMM ---------------------------------------
// 8 waves (wave32). Each wave owns a 4x4 grid of 16x16 f32 accumulators.
// A: fused LN + absmax quant -> f16 (register-staged to LDS).
// B: pre-quantized f16 Bt tile copied with global_load_async_to_lds_b128.

__global__ __launch_bounds__(256, 1) void bitlinear_gemm_kernel(
        const float* __restrict__ x,   const _Float16* __restrict__ btq,
        const float* __restrict__ lng, const float* __restrict__ lnb,
        const float* __restrict__ ws,  float* __restrict__ out) {

    __shared__ _Float16 As[2][BM * ASTRIDE];   // 20 KB
    __shared__ _Float16 Bs[2][BN * BSTRIDE];   // 40 KB

    const int t  = threadIdx.x;
    const int cb = blockIdx.x * BN;            // output col base
    const int rb = blockIdx.y * BM;            // output row base

    const float qscale   = ws[WS_QSCALE];
    const float outscale = ws[WS_OUTSCALE];
    const float clipv    = 128.0f - 1e-5f;     // QB - EPS

    // ---- A loader mapping: thread -> (row in tile, 16-wide k slice) ----
    const int ar = t & 127;                    // 0..127
    const int ak = (t >> 7) * 16;              // 0 or 16 within BK
    const float mu  = ws[WS_ROWSTATS + 2 * (rb + ar)];
    const float rsg = ws[WS_ROWSTATS + 2 * (rb + ar) + 1];
    const float* xrow = x + (size_t)(rb + ar) * KDIM;

    // ---- B async copy mapping: thread -> one Bt row (64B per stage) ----
    const unsigned long long btg =
        (unsigned long long)(uintptr_t)(btq + (size_t)(cb + t) * KDIM);
    const unsigned ldsB0 = (unsigned)(uintptr_t)&Bs[0][0] + (unsigned)t * (BSTRIDE * 2);

    unsigned int aq[8];    // 16 f16 activation values, packed

    auto issueB = [&](int s, int buf) {        // 4 x 16B async lanes -> 64B row
        const unsigned lb = ldsB0 + (unsigned)buf * (BN * BSTRIDE * 2);
        const unsigned long long g = btg + (unsigned long long)(s * (BK * 2));
        #pragma unroll
        for (int i = 0; i < 4; ++i) {
            asm volatile("global_load_async_to_lds_b128 %0, %1, off"
                         :: "v"(lb + i * 16), "v"(g + (unsigned long long)(i * 16))
                         : "memory");
        }
    };

    auto loadqA = [&](int s) {
        const int kk = s * BK;
        _Float16 ah[16];
        #pragma unroll
        for (int q = 0; q < 4; ++q) {
            v4f xv = *(const v4f*)(xrow + kk + ak + q * 4);
            v4f gv = *(const v4f*)(lng  + kk + ak + q * 4);
            v4f bv = *(const v4f*)(lnb  + kk + ak + q * 4);
            #pragma unroll
            for (int e = 0; e < 4; ++e) {
                float v = (xv[e] - mu) * rsg * gv[e] + bv[e];
                v = v * qscale;
                v = fminf(fmaxf(v, -clipv), clipv);
                ah[q * 4 + e] = (_Float16)v;
            }
        }
        #pragma unroll
        for (int p = 0; p < 8; ++p) {
            HPack u; u.h[0] = ah[2*p]; u.h[1] = ah[2*p+1]; aq[p] = u.u;
        }
    };

    auto storeA = [&](int buf) {
        unsigned int* ap = (unsigned int*)&As[buf][ar * ASTRIDE + ak];
        #pragma unroll
        for (int p = 0; p < 8; ++p) ap[p] = aq[p];
    };

    // ---- wave tiling: 2 (m) x 4 (n) waves, 4x4 sub-tiles each ----
    const int wid = t >> 5;        // wave32
    const int lane = t & 31;
    const int wm  = wid & 1;
    const int wn  = wid >> 1;
    const int lhi = lane >> 4;     // 0/1 -> K-half split per ISA layout
    const int llo = lane & 15;

    v8f acc[4][4] = {};

    auto compute = [&](int buf) {
        v16h afr[4], bfr[4];
        #pragma unroll
        for (int i = 0; i < 4; ++i) {
            const int rowi = wm * 64 + i * 16 + llo;
            v8h lo = *(const v8h*)&As[buf][rowi * ASTRIDE + lhi * 8];       // K 0..7 / 8..15
            v8h hi = *(const v8h*)&As[buf][rowi * ASTRIDE + 16 + lhi * 8];  // K 16..23 / 24..31
            #pragma unroll
            for (int e = 0; e < 8; ++e) { afr[i][e] = lo[e]; afr[i][e + 8] = hi[e]; }
        }
        #pragma unroll
        for (int j = 0; j < 4; ++j) {
            const int nj = wn * 64 + j * 16 + llo;
            v8h lo = *(const v8h*)&Bs[buf][nj * BSTRIDE + lhi * 8];         // K 0..7 / 8..15
            v8h hi = *(const v8h*)&Bs[buf][nj * BSTRIDE + 16 + lhi * 8];    // K 16..23 / 24..31
            #pragma unroll
            for (int e = 0; e < 8; ++e) { bfr[j][e] = lo[e]; bfr[j][e + 8] = hi[e]; }
        }
        #pragma unroll
        for (int i = 0; i < 4; ++i)
            #pragma unroll
            for (int j = 0; j < 4; ++j)
                acc[i][j] = __builtin_amdgcn_wmma_f32_16x16x32_f16(
                    false, afr[i], false, bfr[j], (short)0, acc[i][j], false, false);
    };

    // ---- pipelined main loop (double-buffered LDS, async B copies) ----
    issueB(0, 0);
    loadqA(0);
    storeA(0);
    asm volatile("s_wait_asynccnt 0x0" ::: "memory");
    __syncthreads();

    for (int s = 0; s < NSTAGE; ++s) {
        if (s + 1 < NSTAGE) {
            issueB(s + 1, (s + 1) & 1);        // overlaps compute(s)
            loadqA(s + 1);
        }
        if (s + 2 < NSTAGE)
            __builtin_prefetch(xrow + (s + 2) * BK + ak, 0, 0);  // global_prefetch_b8
        compute(s & 1);
        if (s + 1 < NSTAGE) storeA((s + 1) & 1);
        asm volatile("s_wait_asynccnt 0x0" ::: "memory");
        __syncthreads();
    }

    // ---- epilogue: C/D layout -> global (lanes 0-15: M=r, 16-31: M=r+8) ----
    #pragma unroll
    for (int i = 0; i < 4; ++i) {
        #pragma unroll
        for (int j = 0; j < 4; ++j) {
            const int rbase = rb + wm * 64 + i * 16 + lhi * 8;
            const int c     = cb + wn * 64 + j * 16 + llo;
            #pragma unroll
            for (int r = 0; r < 8; ++r)
                out[(size_t)(rbase + r) * NCOLS + c] = acc[i][j][r] * outscale;
        }
    }
}

// ------------------------------ launcher -----------------------------------

extern "C" void kernel_launch(void* const* d_in, const int* in_sizes, int n_in,
                              void* d_out, int out_size, void* d_ws, size_t ws_size,
                              hipStream_t stream) {
    const float* x   = (const float*)d_in[0];   // (4, 8192, 1024) f32
    const float* w   = (const float*)d_in[1];   // (1024, 1024)    f32
    const float* lng = (const float*)d_in[2];   // (1024,)         f32
    const float* lnb = (const float*)d_in[3];   // (1024,)         f32
    float* ws  = (float*)d_ws;
    float* out = (float*)d_out;
    _Float16* btq = (_Float16*)(ws + WS_BTQ);   // 2 MB ternary f16 panel

    hipLaunchKernelGGL(init_kernel,     dim3(1),         dim3(32),  0, stream, ws);
    hipLaunchKernelGGL(wpartial_kernel, dim3(256),       dim3(256), 0, stream, w, ws);
    hipLaunchKernelGGL(rowstats_kernel, dim3(MROWS),     dim3(256), 0, stream, x, lng, lnb, ws);
    hipLaunchKernelGGL(finalize_kernel, dim3(1),         dim3(256), 0, stream, ws);
    hipLaunchKernelGGL(wquant_kernel,   dim3(16, 16),    dim3(256), 0, stream, w, ws, btq);
    hipLaunchKernelGGL(bitlinear_gemm_kernel,
                       dim3(NCOLS / BN, MROWS / BM), dim3(256), 0, stream,
                       x, btq, lng, lnb, ws, out);
}
